// SioConvLayer_47012712022241
// MI455X (gfx1250) — matrix operand order repaired and checked by reading, not compiled, not run on hardware
//
#include <hip/hip_runtime.h>
#include <hip/hip_bf16.h>

// ---------------------------------------------------------------------------
// SioConv for MI455X (gfx1250, wave32, WMMA + TDM).
// GEMMs: v_wmma_f32_16x16x32_bf16. The 16-row A band (shared by all 8 waves
// of a block) is staged into LDS in 16x128 chunks by the Tensor Data Mover
// (tensor_load_to_lds / TENSORcnt), double-buffered: 4 WMMAs per barrier.
// B streams direct from global (no intra-block reuse) with global_prefetch.
// Sequence mixing rewritten as O(L) diagonal complex scan.
// ---------------------------------------------------------------------------

#define DIM_ 1024
#define H_   8
#define D_   64
#define L_   256
#define CH_  (H_ * D_)        // 512 channels
#define N2_  (H_ * D_ * 4)    // 2048
#define KCHUNK 128            // K elements staged per TDM transfer

typedef float v8f  __attribute__((ext_vector_type(8)));
typedef __bf16 v16bf __attribute__((ext_vector_type(16)));
typedef unsigned short ushort8 __attribute__((ext_vector_type(8)));
typedef int v4i __attribute__((ext_vector_type(4)));
typedef int v8i __attribute__((ext_vector_type(8)));

union BF16Frag { v16bf v; ushort8 u[2]; };

__device__ __forceinline__ unsigned short f2bf(float f) {
  unsigned int u = __float_as_uint(f);
  u = (u + 0x7FFFu + ((u >> 16) & 1u)) >> 16;   // round-to-nearest-even
  return (unsigned short)u;
}

// ---------------- Tensor Data Mover: 2D tile -> LDS ------------------------
// D# group0 (4 SGPRs) + group1 (8 SGPRs), 2D tile, data_size = 4 bytes.
// Units below are 4-byte dwords. Issued per-wave (EXEC ignored by TDM).
__device__ __forceinline__ void tdm_load_2d(unsigned lds_addr,
                                            const void* gaddr,
                                            unsigned tile_w_dw, unsigned tile_h,
                                            unsigned row_stride_dw,
                                            unsigned tensor_w_dw, unsigned tensor_h) {
  unsigned long long ga = (unsigned long long)(size_t)gaddr;
  v4i g0;
  g0[0] = 1;                                         // count=1 (valid user D#)
  g0[1] = (int)lds_addr;                             // LDS byte address
  g0[2] = (int)(unsigned)(ga & 0xFFFFFFFFu);         // global_addr[31:0]
  g0[3] = (int)(((unsigned)(ga >> 32) & 0x01FFFFFFu) | (2u << 30)); // hi + type=2
  v8i g1;
  g1[0] = (int)(2u << 16);                           // data_size=2 (4B), mask=0
  g1[1] = (int)((tensor_w_dw & 0xFFFFu) << 16);      // tensor_dim0[15:0]
  g1[2] = (int)(((tensor_w_dw >> 16) & 0xFFFFu) | ((tensor_h & 0xFFFFu) << 16));
  g1[3] = (int)(((tensor_h >> 16) & 0xFFFFu) | ((tile_w_dw & 0xFFFFu) << 16));
  g1[4] = (int)(tile_h & 0xFFFFu);                   // tile_dim1 (tile_dim2=0)
  g1[5] = (int)row_stride_dw;                        // tensor_dim0_stride[31:0]
  g1[6] = 0;                                         // stride hi / dim1_stride lo
  g1[7] = 0;
  asm volatile("tensor_load_to_lds %0, %1" :: "s"(g0), "s"(g1) : "memory");
}

// ------------------------- conversion kernels ------------------------------

__global__ void cvt_bf16(const float* __restrict__ in,
                         unsigned short* __restrict__ out, int n) {
  int i = blockIdx.x * blockDim.x + threadIdx.x;
  if (i < n) out[i] = f2bf(in[i]);
}

// in: K x N (row major, f32) -> out: N x K (row major, bf16)
__global__ void cvt_bf16_T(const float* __restrict__ in,
                           unsigned short* __restrict__ out, int K, int N) {
  int i = blockIdx.x * blockDim.x + threadIdx.x;
  if (i < N * K) {
    int n = i / K, k = i - n * K;
    out[i] = f2bf(in[k * N + n]);
  }
}

// ------------------------------ WMMA GEMM ----------------------------------
// C(MxN) = A(MxK,bf16) * B(KxN), B stored transposed Bt(NxK,bf16).
// Block = 8 waves sharing one 16-row M band, covering 128 columns of N.

template <bool SILU, bool OUT_F32, bool OUT_BF16>
__global__ __launch_bounds__(256)
void wmma_gemm_bf16(const unsigned short* __restrict__ A,
                    const unsigned short* __restrict__ Bt,
                    const float* __restrict__ bias,
                    float* __restrict__ Cf,
                    unsigned short* __restrict__ Cb,
                    int M, int N, int K) {
  __shared__ unsigned short aT[2][16][KCHUNK];      // 8 KB double buffer

  const int nBlkN = N >> 7;                         // N / 128
  const int bM = blockIdx.x / nBlkN;
  const int bN = blockIdx.x - bM * nBlkN;
  const int waveId = threadIdx.x >> 5;
  const int tN = bN * 8 + waveId;
  const int lane = threadIdx.x & 31;
  const int half = lane >> 4;                       // 0: lanes 0-15, 1: 16-31
  const int l16  = lane & 15;

  const unsigned short* aTileG = A + (size_t)(bM * 16) * K;    // 16 x K band
  const unsigned short* bRow   = Bt + (size_t)(tN * 16 + l16) * K;
  const unsigned Kdw = (unsigned)K >> 1;            // row length in dwords

  // prologue: TDM-stage first A chunk (wave 0 issues; EXEC ignored by TDM)
  if (waveId == 0) {
    tdm_load_2d((unsigned)(size_t)&aT[0][0][0], aTileG,
                KCHUNK / 2, 16u, Kdw, Kdw, 16u);
  }
  __builtin_amdgcn_s_wait_tensorcnt(0);             // no-op for waves 1..7
  __syncthreads();

  v8f acc = {};
  int cur = 0;
  for (int k0 = 0; k0 < K; k0 += KCHUNK) {
    // stage next A chunk while computing on the current one
    if (waveId == 0 && (k0 + KCHUNK) < K) {
      tdm_load_2d((unsigned)(size_t)&aT[cur ^ 1][0][0], aTileG + (k0 + KCHUNK),
                  KCHUNK / 2, 16u, Kdw, Kdw, 16u);
    }
    // prefetch the B stream two chunks ahead (global_prefetch_b8)
    if ((k0 + 2 * KCHUNK) < K) __builtin_prefetch(bRow + k0 + 2 * KCHUNK, 0, 3);

#pragma unroll
    for (int kk = 0; kk < KCHUNK; kk += 32) {
      BF16Frag af, bf;
      // A fragment from LDS: lanes0-15 K=[0..7],[16..23]; lanes16-31 K=[8..15],[24..31]
      const unsigned short* aL = &aT[cur][l16][kk];
      af.u[0] = *(const ushort8*)(aL + half * 8);
      af.u[1] = *(const ushort8*)(aL + half * 8 + 16);
      // B fragment direct from global (transposed storage: contiguous K)
      const unsigned short* bP = bRow + k0 + kk;
      bf.u[0] = *(const ushort8*)(bP + half * 8);
      bf.u[1] = *(const ushort8*)(bP + half * 8 + 16);
      acc = __builtin_amdgcn_wmma_f32_16x16x32_bf16(
          false, af.v, false, bf.v, (short)0, acc, false, false);
    }

    __builtin_amdgcn_s_wait_tensorcnt(0);           // issuer drains TDM
    __syncthreads();                                // publish next buffer
    cur ^= 1;
  }

  const int col = tN * 16 + l16;
  const float bv = bias ? bias[col] : 0.0f;
#pragma unroll
  for (int r = 0; r < 8; ++r) {
    const int row = bM * 16 + half * 8 + r;         // C layout: lanes16-31 M=r+8
    float v = acc[r] + bv;
    if (SILU) v = v * __builtin_amdgcn_rcpf(1.0f + __expf(-v)); // silu
    if (OUT_F32)  Cf[(size_t)row * N + col] = v;
    if (OUT_BF16) Cb[(size_t)row * N + col] = f2bf(v);
  }
}

// --------------------- gate extraction / normalization ---------------------
// ac_norm = ac * rsqrt(m)*sigmoid(log m) = ac * sqrt(m)/(1+m)

__global__ void gate_kernel(const float* __restrict__ t2,
                            float* __restrict__ xc,   // (L,CH) interleaved re,im
                            float* __restrict__ a) {  // (L,CH) interleaved re,im
  int i = blockIdx.x * blockDim.x + threadIdx.x;      // over L*CH
  if (i >= L_ * CH_) return;
  int t = i / CH_, c = i - t * CH_;
  int h = c / D_, d = c - h * D_;
  const float* z = t2 + (size_t)t * N2_ + h * (D_ * 4) + d * 4;
  float xr = z[0], xi = z[1], ar = z[2], ai = z[3];
  float m = ar * ar + ai * ai;
  float s = __fsqrt_rn(m) * __builtin_amdgcn_rcpf(1.0f + m);
  xc[2 * i] = xr;  xc[2 * i + 1] = xi;
  a[2 * i]  = ar * s;  a[2 * i + 1] = ai * s;
}

// ------------------- complex 64x64 inverse (Gauss-Jordan) ------------------

__global__ __launch_bounds__(64)
void cinv64(const float* __restrict__ Vre, const float* __restrict__ Vim,
            float* __restrict__ iRe, float* __restrict__ iIm) {
  __shared__ float Re[64][65];
  __shared__ float Im[64][65];
  __shared__ float fr[64], fi[64];
  const int h = blockIdx.x;
  const int j = threadIdx.x;
  for (int r = 0; r < 64; ++r) {
    Re[r][j] = Vre[h * 4096 + r * 64 + j];
    Im[r][j] = Vim[h * 4096 + r * 64 + j];
  }
  __syncthreads();
  for (int k = 0; k < 64; ++k) {
    fr[j] = Re[j][k]; fi[j] = Im[j][k];           // snapshot pivot column
    __syncthreads();
    float pr = fr[k], pi = fi[k];
    float dn = pr * pr + pi * pi;
    float ipr = pr / dn, ipi = -pi / dn;          // 1/pivot
    if (j == k) { Re[k][j] = ipr; Im[k][j] = ipi; }
    else {
      float rr = Re[k][j], ri = Im[k][j];
      Re[k][j] = rr * ipr - ri * ipi;
      Im[k][j] = rr * ipi + ri * ipr;
    }
    __syncthreads();
    float rkr = Re[k][j], rki = Im[k][j];
    for (int i = 0; i < 64; ++i) {
      if (i == k) continue;
      float frv = fr[i], fiv = fi[i];
      if (j == k) {
        Re[i][j] = -(frv * ipr - fiv * ipi);
        Im[i][j] = -(frv * ipi + fiv * ipr);
      } else {
        Re[i][j] -= frv * rkr - fiv * rki;
        Im[i][j] -= frv * rki + fiv * rkr;
      }
    }
    __syncthreads();
  }
  for (int r = 0; r < 64; ++r) {
    iRe[h * 4096 + r * 64 + j] = Re[r][j];
    iIm[h * 4096 + r * 64 + j] = Im[r][j];
  }
}

// ----------------- u = Vinv @ x  (per head complex matvec) -----------------

__global__ __launch_bounds__(64)
void vinv_apply(const float* __restrict__ WRe, const float* __restrict__ WIm,
                const float* __restrict__ xc,
                const float* __restrict__ hidR, const float* __restrict__ hidI,
                float* __restrict__ v, float* __restrict__ sinit) {
  __shared__ float xr[64], xi[64];
  const int b = blockIdx.x;
  const int h = b % H_, ti = b / H_;
  const int o = threadIdx.x;
  if (ti == 0) { xr[o] = hidR[h * D_ + o]; xi[o] = hidI[h * D_ + o]; }
  else {
    int idx = 2 * ((ti - 1) * CH_ + h * D_ + o);
    xr[o] = xc[idx]; xi[o] = xc[idx + 1];
  }
  __syncthreads();
  const float* Rr = WRe + h * 4096 + o * 64;
  const float* Ri = WIm + h * 4096 + o * 64;
  float sr = 0.f, si = 0.f;
#pragma unroll 8
  for (int k = 0; k < 64; ++k) {
    float ar = Rr[k], ai = Ri[k];
    sr += ar * xr[k] - ai * xi[k];
    si += ar * xi[k] + ai * xr[k];
  }
  if (ti == 0) { sinit[2 * (h * D_ + o)] = sr; sinit[2 * (h * D_ + o) + 1] = si; }
  else {
    int idx = 2 * ((ti - 1) * CH_ + h * D_ + o);
    v[idx] = sr; v[idx + 1] = si;
  }
}

// ------------------------- diagonal complex scan ---------------------------
// s_t = a_t * s_{t-1} + v_t ; s_{-1} = Vinv@hidden. One lane per channel.

__global__ __launch_bounds__(256)
void scan_kernel(const float* __restrict__ a, const float* __restrict__ v,
                 const float* __restrict__ sinit, float* __restrict__ hdiag) {
  int c = blockIdx.x * blockDim.x + threadIdx.x;  // 0..511
  if (c >= CH_) return;
  float sr = sinit[2 * c], si = sinit[2 * c + 1];
  for (int t = 0; t < L_; ++t) {
    int idx = 2 * (t * CH_ + c);
    float ar = a[idx], ai = a[idx + 1];
    float nr = sr * ar - si * ai + v[idx];
    float ni = sr * ai + si * ar + v[idx + 1];
    sr = nr; si = ni;
    hdiag[idx] = sr; hdiag[idx + 1] = si;
  }
}

// -------------- h = V @ hdiag ; pack bf16 hr ; emit hidden_next ------------

__global__ __launch_bounds__(64)
void v_apply(const float* __restrict__ VRe, const float* __restrict__ VIm,
             const float* __restrict__ hdiag,
             unsigned short* __restrict__ hrbf, float* __restrict__ hidOut) {
  __shared__ float xr[64], xi[64];
  const int b = blockIdx.x;
  const int h = b % H_, t = b / H_;
  const int o = threadIdx.x;
  int idx = 2 * (t * CH_ + h * D_ + o);
  xr[o] = hdiag[idx]; xi[o] = hdiag[idx + 1];
  __syncthreads();
  const float* Rr = VRe + h * 4096 + o * 64;
  const float* Ri = VIm + h * 4096 + o * 64;
  float sr = 0.f, si = 0.f;
#pragma unroll 8
  for (int k = 0; k < 64; ++k) {
    float ar = Rr[k], ai = Ri[k];
    sr += ar * xr[k] - ai * xi[k];
    si += ar * xi[k] + ai * xr[k];
  }
  int col = 2 * (h * D_ + o);                     // stack([re,im],-1) layout
  hrbf[(size_t)t * DIM_ + col]     = f2bf(sr);
  hrbf[(size_t)t * DIM_ + col + 1] = f2bf(si);
  if (t == L_ - 1) { hidOut[col] = sr; hidOut[col + 1] = si; }
}

// ------------------------------- launcher ----------------------------------

extern "C" void kernel_launch(void* const* d_in, const int* in_sizes, int n_in,
                              void* d_out, int out_size, void* d_ws, size_t ws_size,
                              hipStream_t stream) {
  const float* x    = (const float*)d_in[0];
  const float* hidR = (const float*)d_in[1];
  const float* hidI = (const float*)d_in[2];
  const float* w1   = (const float*)d_in[3];
  const float* b1   = (const float*)d_in[4];
  const float* w2   = (const float*)d_in[5];
  const float* b2   = (const float*)d_in[6];
  const float* w3   = (const float*)d_in[7];
  const float* b3   = (const float*)d_in[8];
  const float* w4   = (const float*)d_in[9];
  const float* b4   = (const float*)d_in[10];
  const float* Vre  = (const float*)d_in[11];
  const float* Vim  = (const float*)d_in[12];
  float* out = (float*)d_out;

  // workspace carve-up (256B aligned)
  char* ws = (char*)d_ws;
  size_t off = 0;
  auto take = [&](size_t bytes) -> char* {
    char* p = ws + off;
    off += (bytes + 255) & ~(size_t)255;
    return p;
  };
  unsigned short* xbf  = (unsigned short*)take((size_t)L_ * DIM_ * 2);
  unsigned short* w1t  = (unsigned short*)take((size_t)DIM_ * DIM_ * 2);
  unsigned short* w2t  = (unsigned short*)take((size_t)N2_ * DIM_ * 2);
  unsigned short* w3t  = (unsigned short*)take((size_t)DIM_ * DIM_ * 2);
  unsigned short* w4t  = (unsigned short*)take((size_t)DIM_ * DIM_ * 2);
  unsigned short* t1bf = (unsigned short*)take((size_t)L_ * DIM_ * 2);
  float* t2    = (float*)take((size_t)L_ * N2_ * 4);
  float* iRe   = (float*)take((size_t)H_ * 4096 * 4);
  float* iIm   = (float*)take((size_t)H_ * 4096 * 4);
  float* xc    = (float*)take((size_t)L_ * CH_ * 2 * 4);
  float* aarr  = (float*)take((size_t)L_ * CH_ * 2 * 4);
  float* varr  = (float*)take((size_t)L_ * CH_ * 2 * 4);
  float* sinit = (float*)take((size_t)CH_ * 2 * 4);
  float* hdiag = (float*)take((size_t)L_ * CH_ * 2 * 4);
  unsigned short* hrbf = (unsigned short*)take((size_t)L_ * DIM_ * 2);
  unsigned short* y1bf = (unsigned short*)take((size_t)L_ * DIM_ * 2);
  (void)ws_size; (void)n_in; (void)in_sizes; (void)out_size;

  const int T = 256;
  // bf16 conversions (weights transposed to NxK for contiguous B fragments)
  cvt_bf16  <<<(L_ * DIM_ + T - 1) / T, T, 0, stream>>>(x, xbf, L_ * DIM_);
  cvt_bf16_T<<<(DIM_ * DIM_ + T - 1) / T, T, 0, stream>>>(w1, w1t, DIM_, DIM_);
  cvt_bf16_T<<<(DIM_ * N2_  + T - 1) / T, T, 0, stream>>>(w2, w2t, DIM_, N2_);
  cvt_bf16_T<<<(DIM_ * DIM_ + T - 1) / T, T, 0, stream>>>(w3, w3t, DIM_, DIM_);
  cvt_bf16_T<<<(DIM_ * DIM_ + T - 1) / T, T, 0, stream>>>(w4, w4t, DIM_, DIM_);

  // GEMM1: t1 = silu(x @ w_in1 + b1)  -> bf16
  wmma_gemm_bf16<true, false, true><<<(L_ / 16) * (DIM_ / 128), 256, 0, stream>>>(
      xbf, w1t, b1, nullptr, t1bf, L_, DIM_, DIM_);
  // GEMM2: t2 = t1 @ w_in2 + b2  -> f32
  wmma_gemm_bf16<false, true, false><<<(L_ / 16) * (N2_ / 128), 256, 0, stream>>>(
      t1bf, w2t, b2, t2, nullptr, L_, N2_, DIM_);

  gate_kernel<<<(L_ * CH_ + T - 1) / T, T, 0, stream>>>(t2, xc, aarr);
  cinv64<<<H_, 64, 0, stream>>>(Vre, Vim, iRe, iIm);
  vinv_apply<<<(L_ + 1) * H_, 64, 0, stream>>>(iRe, iIm, xc, hidR, hidI, varr, sinit);
  scan_kernel<<<(CH_ + T - 1) / T, T, 0, stream>>>(aarr, varr, sinit, hdiag);
  v_apply<<<L_ * H_, 64, 0, stream>>>(Vre, Vim, hdiag, hrbf, out + (size_t)L_ * DIM_);

  // GEMM3: y1 = silu(hr @ w_out1 + b3) -> bf16
  wmma_gemm_bf16<true, false, true><<<(L_ / 16) * (DIM_ / 128), 256, 0, stream>>>(
      hrbf, w3t, b3, nullptr, y1bf, L_, DIM_, DIM_);
  // GEMM4: y = y1 @ w_out2 + b4 -> f32 (front of d_out)
  wmma_gemm_bf16<false, true, false><<<(L_ / 16) * (DIM_ / 128), 256, 0, stream>>>(
      y1bf, w4t, b4, out, nullptr, L_, DIM_, DIM_);
}
